// GNN_48936857371128
// MI455X (gfx1250) — matrix operand order repaired and checked by reading, not compiled, not run on hardware
//
#include <hip/hip_runtime.h>
#include <math.h>

#define HDIM 256
#define GNUM 256

typedef __bf16 bf16_t;
typedef __attribute__((ext_vector_type(16))) __bf16 v16bf;
typedef __attribute__((ext_vector_type(8)))  __bf16 v8bf;
typedef __attribute__((ext_vector_type(8)))  float  v8f;

// ---------------- weight transpose + fp32 -> bf16 (zero-padded K) ----------------
__global__ void k_wt_conv(const float* __restrict__ W, bf16_t* __restrict__ Wt,
                          int K, int Kp) {
  int idx = blockIdx.x * 256 + threadIdx.x;
  if (idx >= HDIM * Kp) return;
  int n = idx / Kp, k = idx - n * Kp;
  float v = (k < K) ? W[(size_t)k * HDIM + n] : 0.0f;
  Wt[idx] = (bf16_t)v;
}

// ---------------- degrees & norms ----------------
__global__ void k_degrees(const int* __restrict__ src, const int* __restrict__ dst,
                          float* deg_out, float* deg_in, int E) {
  int e = blockIdx.x * 256 + threadIdx.x;
  if (e < E) {
    atomicAdd(&deg_out[src[e]], 1.0f);
    atomicAdd(&deg_in[dst[e]], 1.0f);
  }
}

__global__ void k_norms(const float* deg_out, const float* deg_in,
                        float* norm_o, float* norm_i, int N) {
  int v = blockIdx.x * 256 + threadIdx.x;
  if (v < N) {
    float o = deg_out[v], i = deg_in[v];
    norm_o[v] = o > 0.f ? rsqrtf(o) : 0.f;
    norm_i[v] = i > 0.f ? rsqrtf(i) : 0.f;
  }
}

// ---------------- concat [a|c|x] * norm_o -> bf16 Xin [N,32] ----------------
__global__ void k_concat(const float* __restrict__ a_t, const float* __restrict__ c_t,
                         const float* __restrict__ x_t, const float* __restrict__ norm_o,
                         bf16_t* __restrict__ Xin, int N) {
  int t = blockIdx.x * 256 + threadIdx.x;
  int v = t >> 5, c = t & 31;
  if (v >= N) return;
  float val = 0.f;
  if (c < 16)      val = a_t[(size_t)v * 16 + c];
  else if (c < 24) val = c_t[(size_t)v * 8 + (c - 16)];
  else if (c < 27) val = x_t[(size_t)v * 3 + (c - 24)];
  Xin[(size_t)v * 32 + c] = (bf16_t)(val * norm_o[v]);
}

// ---------------- edge features fp32[E,16] -> bf16 [E,32] zero-padded ----------------
__global__ void k_edge_prep(const float* __restrict__ e_t, bf16_t* __restrict__ Ebf, int E) {
  long long t = (long long)blockIdx.x * 256 + threadIdx.x;
  int e = (int)(t >> 5), c = (int)(t & 31);
  if (e >= E) return;
  float v = (c < 16) ? e_t[(size_t)e * 16 + c] : 0.f;
  Ebf[(size_t)e * 32 + c] = (bf16_t)v;
}

// ---------------- bf16 WMMA GEMM: out[rows,256] = A(bf16,[rows,KP]) @ Wt^T ----------------
// Block: 32 rows x 128 cols, 8 waves; wave w owns cols w*16..w*16+15 over all 32 rows
// (two accumulators sharing one B fragment per K-step -> 2 WMMAs per step per wave).
// Entire 32xKP A tile staged to LDS once (single barrier).
// scatter_dst != null: atomicAdd rows into out[scatter_dst[r]] (fused SpMM-by-row).
template <int KP>
__global__ __launch_bounds__(256) void k_gemm_bf16(
    const bf16_t* __restrict__ A, const bf16_t* __restrict__ Wt,
    const float* __restrict__ bias, const int* __restrict__ scatter_dst,
    float* __restrict__ out, int rows) {
  constexpr int LSTR = KP + 8;                       // LDS row stride (bank-safe, 16B aligned)
  __shared__ __align__(16) bf16_t tA[32 * LSTR];
  const int tid  = threadIdx.x;
  const int wave = tid >> 5, lane = tid & 31;
  const int half = lane >> 4, l16 = lane & 15;
  const int rowBase = blockIdx.x * 32;
  const int colBase = blockIdx.y * 128 + wave * 16;

  // ---- stage whole A tile: 32 rows x KP bf16; tail rows clamped (outputs masked) ----
  constexpr int CPR = KP / 4;          // uint2 (4x bf16) chunks per row
  #pragma unroll
  for (int c = tid; c < 32 * CPR; c += 256) {
    int r  = c / CPR;
    int kq = (c - r * CPR) * 4;
    int gr = min(rowBase + r, rows - 1);
    *(uint2*)&tA[r * LSTR + kq] = *(const uint2*)(A + (size_t)gr * KP + kq);
  }
  __syncthreads();

  const bf16_t* __restrict__ bp0 = Wt + (size_t)(colBase + l16) * KP + half * 8;
  v8f acc0 = {0.f, 0.f, 0.f, 0.f, 0.f, 0.f, 0.f, 0.f};
  v8f acc1 = {0.f, 0.f, 0.f, 0.f, 0.f, 0.f, 0.f, 0.f};

  #pragma unroll
  for (int ks = 0; ks < KP / 32; ++ks) {
    // B fragment: lane col = colBase+l16; K = half*8 + {0..7} and 16 + half*8 + {0..7}
    const bf16_t* bp = bp0 + ks * 32;
    v8bf blo = *(const v8bf*)bp;
    v8bf bhi = *(const v8bf*)(bp + 16);
    v16bf bfrag = __builtin_shufflevector(blo, bhi, 0,1,2,3,4,5,6,7,8,9,10,11,12,13,14,15);
    // A fragments: lane row = l16 (strip 0) and 16+l16 (strip 1);
    // K = half*8 + {0..7} and 16 + half*8 + {0..7}
    const int kb = ks * 32 + half * 8;
    v8bf a0lo = *(const v8bf*)&tA[l16 * LSTR + kb];
    v8bf a0hi = *(const v8bf*)&tA[l16 * LSTR + kb + 16];
    v16bf a0 = __builtin_shufflevector(a0lo, a0hi, 0,1,2,3,4,5,6,7,8,9,10,11,12,13,14,15);
    v8bf a1lo = *(const v8bf*)&tA[(16 + l16) * LSTR + kb];
    v8bf a1hi = *(const v8bf*)&tA[(16 + l16) * LSTR + kb + 16];
    v16bf a1 = __builtin_shufflevector(a1lo, a1hi, 0,1,2,3,4,5,6,7,8,9,10,11,12,13,14,15);
    acc0 = __builtin_amdgcn_wmma_f32_16x16x32_bf16(false, a0, false, bfrag,
                                                   (short)0, acc0, false, false);
    acc1 = __builtin_amdgcn_wmma_f32_16x16x32_bf16(false, a1, false, bfrag,
                                                   (short)0, acc1, false, false);
  }

  const int oc = colBase + l16;
  const float b = bias ? bias[oc] : 0.0f;
  const int rA = rowBase + half * 8;        // C layout: VGPR v -> M = v + 8*half
  const int rB = rA + 16;
  const bool full = (rowBase + 32 <= rows); // scalar-uniform fast path (no tail guards)
  if (scatter_dst) {
    if (full) {
      #pragma unroll
      for (int v = 0; v < 8; ++v)
        atomicAdd(&out[(size_t)scatter_dst[rA + v] * HDIM + oc], acc0[v] + b);
      #pragma unroll
      for (int v = 0; v < 8; ++v)
        atomicAdd(&out[(size_t)scatter_dst[rB + v] * HDIM + oc], acc1[v] + b);
    } else {
      #pragma unroll
      for (int v = 0; v < 8; ++v) {
        int r = rA + v;
        if (r < rows) atomicAdd(&out[(size_t)scatter_dst[r] * HDIM + oc], acc0[v] + b);
      }
      #pragma unroll
      for (int v = 0; v < 8; ++v) {
        int r = rB + v;
        if (r < rows) atomicAdd(&out[(size_t)scatter_dst[r] * HDIM + oc], acc1[v] + b);
      }
    }
  } else {
    if (full) {
      #pragma unroll
      for (int v = 0; v < 8; ++v) out[(size_t)(rA + v) * HDIM + oc] = acc0[v] + b;
      #pragma unroll
      for (int v = 0; v < 8; ++v) out[(size_t)(rB + v) * HDIM + oc] = acc1[v] + b;
    } else {
      #pragma unroll
      for (int v = 0; v < 8; ++v) {
        int r = rA + v;
        if (r < rows) out[(size_t)r * HDIM + oc] = acc0[v] + b;
      }
      #pragma unroll
      for (int v = 0; v < 8; ++v) {
        int r = rB + v;
        if (r < rows) out[(size_t)r * HDIM + oc] = acc1[v] + b;
      }
    }
  }
}

// ---------------- SpMM scatter: acc[dst[e]] += t[src[e]] ----------------
__global__ void k_scatter(const float* __restrict__ t, const int* __restrict__ src,
                          const int* __restrict__ dst, float* __restrict__ acc, int E) {
  long long tid = (long long)blockIdx.x * 256 + threadIdx.x;
  int e = (int)(tid >> 6), l = (int)(tid & 63);
  if (e >= E) return;
  int s = src[e], d = dst[e];
  const float4 v = *(const float4*)(t + (size_t)s * HDIM + l * 4);
  float* o = acc + (size_t)d * HDIM + l * 4;
  atomicAdd(o + 0, v.x); atomicAdd(o + 1, v.y);
  atomicAdd(o + 2, v.z); atomicAdd(o + 3, v.w);
}

// ---------------- block-wide (256 thread) sum via wave shuffle + LDS ----------------
__device__ __forceinline__ float block_sum256(float v) {
  __shared__ float red[8];
  #pragma unroll
  for (int o = 16; o > 0; o >>= 1) v += __shfl_down(v, o, 32);
  int wave = threadIdx.x >> 5, lane = threadIdx.x & 31;
  if (lane == 0) red[wave] = v;
  __syncthreads();
  if (threadIdx.x == 0) {
    float s = 0.f;
    #pragma unroll
    for (int i = 0; i < 8; ++i) s += red[i];
    red[0] = s;
  }
  __syncthreads();
  float r = red[0];
  __syncthreads();
  return r;
}

// ------- epilogue: m*norm_i + b [+ emean] [-> LN -> SiLU] [+ res]
//         outf (fp32, optional) and outbf = bf16(x * norm_o) for the next GEMM -------
__global__ __launch_bounds__(256) void k_epilogue(
    const float* __restrict__ m, const float* __restrict__ norm_i,
    const float* __restrict__ norm_o, const float* __restrict__ bias,
    const float* __restrict__ esum, const float* __restrict__ deg_in,
    const float* __restrict__ gamma, const float* __restrict__ beta,
    const float* __restrict__ res, float* __restrict__ outf,
    bf16_t* __restrict__ outbf, int do_ln_silu) {
  int v = blockIdx.x, c = threadIdx.x;
  size_t idx = (size_t)v * HDIM + c;
  float x = m[idx] * norm_i[v] + bias[c];
  if (esum) x += esum[idx] / fmaxf(deg_in[v], 1.0f);
  if (do_ln_silu) {
    float mu  = block_sum256(x) * (1.0f / HDIM);
    float d   = x - mu;
    float var = block_sum256(d * d) * (1.0f / HDIM);
    float xn  = d * rsqrtf(var + 1e-5f) * gamma[c] + beta[c];
    x = xn / (1.0f + __expf(-xn));  // SiLU
  }
  if (res) x += res[idx];
  if (outf) outf[idx] = x;
  outbf[idx] = (bf16_t)(x * norm_o[v]);
}

// ---------------- graph mean-pool (atomic) + head ----------------
__global__ void k_pool(const float* __restrict__ h, const int* __restrict__ gid,
                       float* pooled, float* cnt, int N) {
  long long tid = (long long)blockIdx.x * 256 + threadIdx.x;
  int v = (int)(tid >> 6), l = (int)(tid & 63);
  if (v >= N) return;
  int g = gid[v];
  if (l == 0) atomicAdd(&cnt[g], 1.0f);
  const float4 x = *(const float4*)(h + (size_t)v * HDIM + l * 4);
  float* o = pooled + (size_t)g * HDIM + l * 4;
  atomicAdd(o + 0, x.x); atomicAdd(o + 1, x.y);
  atomicAdd(o + 2, x.z); atomicAdd(o + 3, x.w);
}

__global__ __launch_bounds__(256) void k_head(const float* __restrict__ pooled,
                                              const float* __restrict__ cnt,
                                              const float* __restrict__ W_head,
                                              const float* __restrict__ b_head,
                                              float* __restrict__ out) {
  int g = blockIdx.x, c = threadIdx.x;
  float inv = 1.0f / fmaxf(cnt[g], 1.0f);
  float v = pooled[(size_t)g * HDIM + c] * inv * W_head[c];
  float s = block_sum256(v);
  if (c == 0) {
    float x = s + b_head[0];
    out[g] = (x > 20.f) ? x : log1pf(__expf(x));  // softplus
  }
}

// ---------------- orchestration ----------------
extern "C" void kernel_launch(void* const* d_in, const int* in_sizes, int n_in,
                              void* d_out, int out_size, void* d_ws, size_t ws_size,
                              hipStream_t stream) {
  const float* a_t   = (const float*)d_in[0];
  const float* c_t   = (const float*)d_in[1];
  const float* x_t   = (const float*)d_in[2];
  const float* e_t   = (const float*)d_in[3];
  const int*   src   = (const int*)d_in[4];
  const int*   dst   = (const int*)d_in[5];
  const int*   gid   = (const int*)d_in[6];
  const float* W_in  = (const float*)d_in[7];
  const float* b_in  = (const float*)d_in[8];
  const float* W_e   = (const float*)d_in[9];
  const float* b_e   = (const float*)d_in[10];
  const float* Wc1   = (const float*)d_in[11];
  const float* bc1   = (const float*)d_in[12];
  const float* g1    = (const float*)d_in[13];
  const float* be1   = (const float*)d_in[14];
  const float* Wc2   = (const float*)d_in[15];
  const float* bc2   = (const float*)d_in[16];
  const float* g2    = (const float*)d_in[17];
  const float* be2   = (const float*)d_in[18];
  const float* W_hd  = (const float*)d_in[19];
  const float* b_hd  = (const float*)d_in[20];
  (void)n_in; (void)out_size; (void)ws_size;

  const int N     = in_sizes[0] / 16;
  const int E     = in_sizes[4];
  const int DEPTH = in_sizes[11] / (HDIM * HDIM);

  char* p = (char*)d_ws;
  auto carve = [&](size_t b) -> char* {
    char* r = p; p += (b + 255) & ~(size_t)255; return r;
  };
  float*  deg_out = (float*)carve((size_t)N * 4);
  float*  deg_in  = (float*)carve((size_t)N * 4);
  float*  norm_o  = (float*)carve((size_t)N * 4);
  float*  norm_i  = (float*)carve((size_t)N * 4);
  float*  hb      = (float*)carve((size_t)N * HDIM * 4);   // fp32 state (residual)
  float*  t1      = (float*)carve((size_t)N * HDIM * 4);   // scratch fp32
  float*  t2      = (float*)carve((size_t)N * HDIM * 4);   // scratch fp32
  bf16_t* Abf     = (bf16_t*)carve((size_t)N * HDIM * 2);  // pre-scaled bf16 GEMM input
  bf16_t* Xin     = (bf16_t*)carve((size_t)N * 32 * 2);    // padded scaled input feats
  bf16_t* Ebf     = (bf16_t*)carve((size_t)E * 32 * 2);    // padded edge feats
  float*  pooled  = (float*)carve((size_t)GNUM * HDIM * 4);
  float*  cnt     = (float*)carve((size_t)GNUM * 4);
  bf16_t* Wt_in   = (bf16_t*)carve((size_t)HDIM * 32 * 2);
  bf16_t* Wt_e    = (bf16_t*)carve((size_t)HDIM * 32 * 2);
  bf16_t* Wt1     = (bf16_t*)carve((size_t)DEPTH * HDIM * HDIM * 2);
  bf16_t* Wt2     = (bf16_t*)carve((size_t)DEPTH * HDIM * HDIM * 2);

  const size_t hbytes = (size_t)N * HDIM * 4;
  dim3 blk(256);
  dim3 gN((N + 31) / 32, 2);
  dim3 gE((E + 31) / 32, 2);
  int scatterBlocks = (int)(((size_t)E * 64 + 255) / 256);

  // Weight prep
  int wb32 = (HDIM * 32 + 255) / 256;
  int wbH  = (HDIM * HDIM + 255) / 256;
  k_wt_conv<<<wb32, blk, 0, stream>>>(W_in, Wt_in, 27, 32);
  k_wt_conv<<<wb32, blk, 0, stream>>>(W_e, Wt_e, 16, 32);
  for (int l = 0; l < DEPTH; ++l) {
    k_wt_conv<<<wbH, blk, 0, stream>>>(Wc1 + (size_t)l * HDIM * HDIM,
                                       Wt1 + (size_t)l * HDIM * HDIM, HDIM, HDIM);
    k_wt_conv<<<wbH, blk, 0, stream>>>(Wc2 + (size_t)l * HDIM * HDIM,
                                       Wt2 + (size_t)l * HDIM * HDIM, HDIM, HDIM);
  }

  // Degrees / norms / feature prep
  hipMemsetAsync(deg_out, 0, (size_t)N * 4, stream);
  hipMemsetAsync(deg_in,  0, (size_t)N * 4, stream);
  k_degrees<<<(E + 255) / 256, blk, 0, stream>>>(src, dst, deg_out, deg_in, E);
  k_norms<<<(N + 255) / 256, blk, 0, stream>>>(deg_out, deg_in, norm_o, norm_i, N);
  k_concat<<<(int)(((size_t)N * 32 + 255) / 256), blk, 0, stream>>>(a_t, c_t, x_t, norm_o, Xin, N);
  k_edge_prep<<<(int)(((size_t)E * 32 + 255) / 256), blk, 0, stream>>>(e_t, Ebf, E);

  // Edge linear fused with segment-sum by dst: t2 = sum_dst(e_t@W_e + b_e)
  hipMemsetAsync(t2, 0, hbytes, stream);
  k_gemm_bf16<32><<<gE, blk, 0, stream>>>(Ebf, Wt_e, b_e, dst, t2, E);

  // Input gconv: hb = Xin@W_in ; t1 = scatter(hb) ; epilogue -> hb fp32 + Abf bf16
  k_gemm_bf16<32><<<gN, blk, 0, stream>>>(Xin, Wt_in, nullptr, nullptr, hb, N);
  hipMemsetAsync(t1, 0, hbytes, stream);
  k_scatter<<<scatterBlocks, blk, 0, stream>>>(hb, src, dst, t1, E);
  k_epilogue<<<N, blk, 0, stream>>>(t1, norm_i, norm_o, b_in, t2, deg_in,
                                    nullptr, nullptr, nullptr, hb, Abf, 0);

  for (int l = 0; l < DEPTH; ++l) {
    const size_t wl = (size_t)l * HDIM * HDIM;
    const size_t bl = (size_t)l * HDIM;
    // t = silu(ln(gconv(h, W1)))  (fp32 copy not needed; only bf16 feeds next GEMM)
    k_gemm_bf16<HDIM><<<gN, blk, 0, stream>>>(Abf, Wt1 + wl, nullptr, nullptr, t1, N);
    hipMemsetAsync(t2, 0, hbytes, stream);
    k_scatter<<<scatterBlocks, blk, 0, stream>>>(t1, src, dst, t2, E);
    k_epilogue<<<N, blk, 0, stream>>>(t2, norm_i, norm_o, bc1 + bl, nullptr, nullptr,
                                      g1 + bl, be1 + bl, nullptr, nullptr, Abf, 1);
    // h = silu(ln(gconv(t, W2))) + h
    k_gemm_bf16<HDIM><<<gN, blk, 0, stream>>>(Abf, Wt2 + wl, nullptr, nullptr, t2, N);
    hipMemsetAsync(t1, 0, hbytes, stream);
    k_scatter<<<scatterBlocks, blk, 0, stream>>>(t2, src, dst, t1, E);
    k_epilogue<<<N, blk, 0, stream>>>(t1, norm_i, norm_o, bc2 + bl, nullptr, nullptr,
                                      g2 + bl, be2 + bl, hb, hb, Abf, 1);
  }

  // Mean pool per graph + softplus head
  hipMemsetAsync(pooled, 0, (size_t)GNUM * HDIM * 4, stream);
  hipMemsetAsync(cnt,    0, (size_t)GNUM * 4, stream);
  k_pool<<<(int)(((size_t)N * 64 + 255) / 256), blk, 0, stream>>>(hb, gid, pooled, cnt, N);
  k_head<<<GNUM, blk, 0, stream>>>(pooled, cnt, W_hd, b_hd, (float*)d_out);
}
